// PixelContrastCrossEntropyLoss_6657199309058
// MI455X (gfx1250) — matrix operand order repaired and checked by reading, not compiled, not run on hardware
//
#include <hip/hip_runtime.h>
#include <hip/hip_bf16.h>
#include <math.h>

// ---------- problem constants (mirroring the reference) ----------
#define NUM_CLASSES 8
#define MAX_SAMPLES 1024
#define MAX_VIEWS   100
#define NROWS       8192      // NUM_CLASSES * MAX_SAMPLES
#define CCH         128       // embedding channels
#define PTOT        262144    // 2*256*512 pixels
#define HW          131072    // 256*512
#define NB          256       // histogram/scatter chunks
#define CHUNK       (PTOT/NB) // 1024 pixels per chunk
#define SCALE       7.0f      // BASE_T / T^2 = 0.07 / 0.01

typedef __attribute__((ext_vector_type(16))) __bf16 v16bf;
typedef __attribute__((ext_vector_type(8)))  __bf16 v8bf;
typedef __attribute__((ext_vector_type(8)))  float  v8f;

static __device__ __forceinline__ unsigned short f2bf(float f) {
    unsigned u = __float_as_uint(f);
    u += ((u >> 16) & 1u) + 0x7FFFu;          // round-to-nearest-even
    return (unsigned short)(u >> 16);
}
static __device__ __forceinline__ float bf2f(unsigned short h) {
    return __uint_as_float(((unsigned)h) << 16);
}

// ---------- 0: zero the bookkeeping region of the workspace ----------
__global__ void k_zero(unsigned* __restrict__ p, int n) {
    int i = blockIdx.x * blockDim.x + threadIdx.x;
    if (i < n) p[i] = 0u;
}

// ---------- 1: per-chunk per-class histogram (one wave32 per chunk) ----------
__global__ void k_hist(const long long* __restrict__ lab, unsigned* __restrict__ hist) {
    int lane = threadIdx.x;                    // blockDim = 32
    int base = blockIdx.x * CHUNK;
    unsigned cnt[NUM_CLASSES];
#pragma unroll
    for (int k = 0; k < NUM_CLASSES; k++) cnt[k] = 0u;
    for (int t = 0; t < CHUNK / 32; t++) {
        int c = (int)lab[base + t * 32 + lane];
#pragma unroll
        for (int k = 0; k < NUM_CLASSES; k++) cnt[k] += (c == k);
    }
#pragma unroll
    for (int k = 0; k < NUM_CLASSES; k++) {
        unsigned v = cnt[k];
        for (int m = 16; m > 0; m >>= 1) v += __shfl_xor(v, m, 32);
        if (lane == 0) hist[blockIdx.x * NUM_CLASSES + k] = v;
    }
}

// ---------- 2: exclusive scan of chunk histograms (8 classes in parallel) ----------
__global__ void k_scan(const unsigned* __restrict__ hist, unsigned* __restrict__ offs) {
    int c = threadIdx.x;
    if (c < NUM_CLASSES) {
        unsigned run = 0;
        for (int b = 0; b < NB; b++) {
            offs[b * NUM_CLASSES + c] = run;
            run += hist[b * NUM_CLASSES + c];
        }
    }
}

// ---------- 3: scatter first MAX_SAMPLES scan-order pixels per class ----------
__global__ void k_scatter(const long long* __restrict__ lab,
                          const unsigned* __restrict__ offs,
                          unsigned* __restrict__ sidx) {
    int lane = threadIdx.x;                    // blockDim = 32
    int base = blockIdx.x * CHUNK;
    unsigned run[NUM_CLASSES];
#pragma unroll
    for (int k = 0; k < NUM_CLASSES; k++) run[k] = offs[blockIdx.x * NUM_CLASSES + k];
    unsigned ltmask = (1u << lane) - 1u;
    for (int t = 0; t < CHUNK / 32; t++) {
        int p = base + t * 32 + lane;
        int c = (int)lab[p];
#pragma unroll
        for (int k = 0; k < NUM_CLASSES; k++) {
            unsigned m = (unsigned)__ballot(c == k);   // wave32: low 32 bits
            if (c == k) {
                unsigned r = run[k] + (unsigned)__popc(m & ltmask);
                if (r < MAX_SAMPLES) sidx[k * MAX_SAMPLES + r] = (unsigned)p;
            }
            run[k] += (unsigned)__popc(m);
        }
    }
}

// ---------- 4: gather + L2-normalize + bf16 convert (one wave per row) ----------
__global__ void k_norm(const float* __restrict__ emb,
                       const unsigned* __restrict__ sidx,
                       unsigned short* __restrict__ Ebf) {
    int lane = threadIdx.x & 31;
    int wv   = threadIdx.x >> 5;
    int i    = blockIdx.x * 8 + wv;            // row 0..8191
    unsigned p = sidx[i];
    int bimg = (int)(p / HW);
    int hw   = (int)(p % HW);
    const float* src = emb + (size_t)bimg * CCH * HW + hw;   // + c*HW per channel
    float v[4];
    float s = 0.f;
#pragma unroll
    for (int t = 0; t < 4; t++) {
        int c = lane + t * 32;
        v[t] = src[(size_t)c * HW];
        s += v[t] * v[t];
    }
    for (int m = 16; m > 0; m >>= 1) s += __shfl_xor(s, m, 32);
    float inv = 1.0f / fmaxf(sqrtf(s), 1e-12f);
#pragma unroll
    for (int t = 0; t < 4; t++)
        Ebf[(size_t)i * CCH + lane + t * 32] = f2bf(v[t] * inv);
}

// ---------- 5: WMMA GEMM  S = E*E^T, fold exp(7*S) into column sums ----------
// Block = 256 threads = 8 waves as 2(M) x 4(N); wave tile = 16(M) x 64(N):
// one A fragment reused across 4 B tiles -> 2.5 b128 loads per WMMA.
__global__ void k_gemm_colsum(const unsigned short* __restrict__ Ebf_us,
                              float* __restrict__ colsum) {
    const __bf16* E = reinterpret_cast<const __bf16*>(Ebf_us);
    __shared__ float cs[256];
    int tid  = threadIdx.x;
    int lane = tid & 31;
    int wv   = tid >> 5;
    int msub = wv & 1;                 // 2 M sub-tiles
    int nsub = wv >> 1;                // 4 N sub-tiles (each 64 wide)
    int m0 = blockIdx.x * 32  + msub * 16;
    int n0 = blockIdx.y * 256 + nsub * 64;
    cs[tid] = 0.f;
    __syncthreads();

    int half = lane >> 4;              // 0: lanes 0-15, 1: lanes 16-31
    int lr   = lane & 15;
    const __bf16* arow = E + (size_t)(m0 + lr) * CCH;
    const __bf16* brow[4];
#pragma unroll
    for (int t = 0; t < 4; t++)
        brow[t] = E + (size_t)(n0 + t * 16 + lr) * CCH;   // B = E^T column == E row
    __builtin_prefetch(arow, 0, 1);
    __builtin_prefetch(brow[0], 0, 1);

    v8f acc[4] = {};
#pragma unroll
    for (int k = 0; k < CCH; k += 32) {
        // A 16x32 bf16: elems 0..7 = K[k+8h .. +7], elems 8..15 = K[k+16+8h .. +7]
        v8bf a0 = *reinterpret_cast<const v8bf*>(arow + k + half * 8);
        v8bf a1 = *reinterpret_cast<const v8bf*>(arow + k + 16 + half * 8);
        v16bf a = __builtin_shufflevector(a0, a1,
                    0,1,2,3,4,5,6,7,8,9,10,11,12,13,14,15);
#pragma unroll
        for (int t = 0; t < 4; t++) {
            // B 32x16 bf16: elems 0..7 = K[k+8h .. +7], elems 8..15 = K[k+16+8h .. +7] for column n
            v8bf b0 = *reinterpret_cast<const v8bf*>(brow[t] + k + half * 8);
            v8bf b1 = *reinterpret_cast<const v8bf*>(brow[t] + k + 16 + half * 8);
            v16bf b = __builtin_shufflevector(b0, b1,
                    0,1,2,3,4,5,6,7,8,9,10,11,12,13,14,15);
            acc[t] = __builtin_amdgcn_wmma_f32_16x16x32_bf16(
                false, a, false, b, (short)0, acc[t], false, false);
        }
    }

    // lane holds column n_g, rows m0 + r + 8*half (r = 0..7) for each sub-tile
#pragma unroll
    for (int t = 0; t < 4; t++) {
        int n_g = n0 + t * 16 + lr;
        float part = 0.f;
#pragma unroll
        for (int r = 0; r < 8; r++) {
            int m_g = m0 + r + half * 8;
            part += (m_g == n_g) ? 0.f : __expf(SCALE * acc[t][r]);  // diag suppressed
        }
        part += __shfl_xor(part, 16, 32);      // combine the two 8-row halves
        if (half == 0) atomicAdd(&cs[nsub * 64 + t * 16 + lr], part);
    }
    __syncthreads();
    atomicAdd(&colsum[blockIdx.y * 256 + tid], cs[tid]);
}

// ---------- 6: scalar positive-sum  s_pos = sum_i sum_j exp(7*<e_i, e_pc>) ----------
__global__ void k_spos(const unsigned short* __restrict__ Ebf,
                       float* __restrict__ spos) {
    int lane = threadIdx.x & 31;
    int wv   = threadIdx.x >> 5;
    int i    = blockIdx.x * 8 + wv;
    const unsigned short* ei = Ebf + (size_t)i * CCH;
    float e[4];
#pragma unroll
    for (int t = 0; t < 4; t++) e[t] = bf2f(ei[lane + t * 32]);
    int bs = (i >> 10) << 10;                  // class block start
    float acc = 0.f;
    for (int j = 0; j < MAX_VIEWS; j++) {
        int pc = bs + ((i * 7 + j * 131) & (MAX_SAMPLES - 1));  // deterministic pick
        const unsigned short* ep = Ebf + (size_t)pc * CCH;
        float d = 0.f;
#pragma unroll
        for (int t = 0; t < 4; t++) d += e[t] * bf2f(ep[lane + t * 32]);
        for (int m = 16; m > 0; m >>= 1) d += __shfl_xor(d, m, 32);
        if (lane == 0 && pc != i) acc += __expf(SCALE * d);     // self-col: exp(-huge)=0
    }
    if (lane == 0) atomicAdd(spos, acc);
}

// ---------- 7: loss = -log(s_pos) + mean_i log(col_sum_i) ----------
__global__ void k_final(const float* __restrict__ colsum,
                        const float* __restrict__ spos,
                        float* __restrict__ out) {
    __shared__ float red[256];
    float s = 0.f;
    for (int i = threadIdx.x; i < NROWS; i += 256) s += logf(colsum[i]);
    red[threadIdx.x] = s;
    __syncthreads();
    for (int step = 128; step > 0; step >>= 1) {
        if (threadIdx.x < step) red[threadIdx.x] += red[threadIdx.x + step];
        __syncthreads();
    }
    if (threadIdx.x == 0) out[0] = -logf(spos[0]) + red[0] / (float)NROWS;
}

extern "C" void kernel_launch(void* const* d_in, const int* in_sizes, int n_in,
                              void* d_out, int out_size, void* d_ws, size_t ws_size,
                              hipStream_t stream) {
    const float*     emb = (const float*)d_in[0];       // (2,128,256,512) f32
    const long long* lab = (const long long*)d_in[1];   // (2,256,512) int64
    float*           out = (float*)d_out;               // scalar f32 loss

    // workspace layout (bytes)
    char* ws = (char*)d_ws;
    float*          colsum = (float*)(ws + 0);          //  32768
    float*          spos   = (float*)(ws + 32768);      //      4 (padded)
    unsigned*       hist   = (unsigned*)(ws + 33024);   //   8192
    unsigned*       offs   = (unsigned*)(ws + 41216);   //   8192
    unsigned*       sidx   = (unsigned*)(ws + 49408);   //  32768
    unsigned short* Ebf    = (unsigned short*)(ws + 82432); // 8192*128*2 = 2 MB

    // zero colsum/spos/hist/offs/sidx (first 82176 bytes = 20544 dwords)
    {
        int n = 82176 / 4;
        k_zero<<<(n + 255) / 256, 256, 0, stream>>>((unsigned*)ws, n);
    }
    k_hist   <<<NB, 32, 0, stream>>>(lab, hist);
    k_scan   <<<1, 32, 0, stream>>>(hist, offs);
    k_scatter<<<NB, 32, 0, stream>>>(lab, offs, sidx);
    k_norm   <<<NROWS / 8, 256, 0, stream>>>(emb, sidx, Ebf);
    k_gemm_colsum<<<dim3(NROWS / 32, NROWS / 256), 256, 0, stream>>>(Ebf, colsum);
    k_spos   <<<NROWS / 8, 256, 0, stream>>>(Ebf, spos);
    k_final  <<<1, 256, 0, stream>>>(colsum, spos, out);
    (void)in_sizes; (void)n_in; (void)out_size; (void)ws_size;
}
